// SimpleALIFRNNTbptt_89756226552396
// MI455X (gfx1250) — matrix operand order repaired and checked by reading, not compiled, not run on hardware
//
#include <hip/hip_runtime.h>
#include <hip/hip_bf16.h>
#include <math.h>

typedef __bf16 bf16;
typedef __attribute__((ext_vector_type(16))) __bf16 v16bf;
typedef __attribute__((ext_vector_type(8)))  float  v8f;

#define T_STEPS 1000
#define BSZ     128
#define IN_DIM  700
#define H_DIM   256
#define OUT_DIM 20
#define KTOT    956      // IN_DIM + H_DIM
#define KPAD    704      // IN_DIM padded to multiple of 32
#define XPITCH  712      // LDS row pitch for x tile (16B aligned, conflict-breaking)
#define WRPITCH 264      // LDS row pitch for Wr / z (16B aligned, 2-way max conflict)
#define OPAD    32       // OUT_DIM padded
#define B0C     0.01f
#define BETA    1.8f

// ---------------------------------------------------------------------------
// Kernel 1: build K-major zero-padded bf16 weight layouts + decay factors.
//   WxT[704][256] : WxT[k][h] = k<700 ? Wh[h][k]      : 0
//   WrT[256][256] : WrT[k][h] =         Wh[h][700+k]
//   WoT[256][32]  : WoT[k][o] = o<20  ? Wo[o][k]      : 0
// ---------------------------------------------------------------------------
__global__ void alif_init_kernel(const float* __restrict__ Wh,
                                 const float* __restrict__ Wo,
                                 const float* __restrict__ tau_mem,
                                 const float* __restrict__ tau_adp,
                                 const float* __restrict__ out_tau,
                                 bf16* __restrict__ WxT,
                                 bf16* __restrict__ WrT,
                                 bf16* __restrict__ WoT,
                                 float* __restrict__ alpha,
                                 float* __restrict__ rho,
                                 float* __restrict__ oalpha,
                                 float* __restrict__ out_tail) {
  int stride = gridDim.x * blockDim.x;
  int tid = blockIdx.x * blockDim.x + threadIdx.x;
  for (int i = tid; i < KPAD * H_DIM; i += stride) {
    int k = i >> 8, h = i & 255;
    WxT[i] = (k < IN_DIM) ? (bf16)Wh[h * KTOT + k] : (bf16)0.0f;
  }
  for (int i = tid; i < H_DIM * H_DIM; i += stride) {
    int k = i >> 8, h = i & 255;
    WrT[i] = (bf16)Wh[h * KTOT + IN_DIM + k];
  }
  for (int i = tid; i < H_DIM * OPAD; i += stride) {
    int k = i >> 5, o = i & 31;
    WoT[i] = (o < OUT_DIM) ? (bf16)Wo[o * H_DIM + k] : (bf16)0.0f;
  }
  for (int i = tid; i < H_DIM; i += stride) {
    alpha[i] = __expf(-1.0f / tau_mem[i]);
    rho[i]   = __expf(-1.0f / tau_adp[i]);
  }
  for (int i = tid; i < OUT_DIM; i += stride) oalpha[i] = __expf(-1.0f / out_tau[i]);
  if (tid < 2) out_tail[tid] = 0.0f;
}

// ---------------------------------------------------------------------------
// Kernel 2: Icur[t,b,h] = sum_k x[t,b,k] * Wh[h,k]   (k < 700)
// One block = one 16-row m-tile; x tile staged in LDS as zero-padded bf16
// (no guards in WMMA loop); 8 waves x 2 n-tiles; B loads k-strided per lane.
// ---------------------------------------------------------------------------
__global__ void __launch_bounds__(256)
alif_input_gemm_kernel(const float* __restrict__ x,
                       const bf16* __restrict__ WxT,
                       float* __restrict__ Icur) {
  __shared__ bf16 x_sh[16][XPITCH];

  int tid  = threadIdx.x;
  int lane = tid & 31;
  int wave = tid >> 5;
  int row0 = blockIdx.x * 16;
  int arow = lane & 15;
  int hi   = lane >> 4;
  int ccol = lane & 15;

  // Stage + convert + zero-pad the 16x700 x tile (coalesced over k).
  for (int i = tid; i < 16 * XPITCH; i += 256) {
    int r = i / XPITCH, k = i - r * XPITCH;
    x_sh[r][k] = (k < IN_DIM) ? (bf16)x[(size_t)(row0 + r) * IN_DIM + k] : (bf16)0.0f;
  }
  __syncthreads();

#pragma unroll
  for (int s = 0; s < 2; ++s) {
    int n0 = (wave * 2 + s) * 16;
    v8f acc = {};
    for (int kc = 0; kc < KPAD / 32; ++kc) {   // 22 full chunks, no guards
      int kb = kc * 32;
      v16bf a, b;
#pragma unroll
      for (int e = 0; e < 16; ++e) {
        int vg = e >> 1, pos = e & 1;
        int k = kb + ((vg & 4) ? 16 : 0) + (hi ? 8 : 0) + ((vg & 3) << 1) + pos;
        a[e] = x_sh[arow][k];
      }
#pragma unroll
      for (int e = 0; e < 16; ++e)             // k-strided gather, column n0+ccol
        b[e] = WxT[(size_t)(kb + ((e & 8) ? 16 : 0) + (hi ? 8 : 0) + (e & 7)) * H_DIM + n0 + ccol];
      acc = __builtin_amdgcn_wmma_f32_16x16x32_bf16(false, a, false, b,
                                                    (short)0, acc, false, false);
    }
#pragma unroll
    for (int e = 0; e < 8; ++e) {
      int crow = (hi ? 8 : 0) + e;
      Icur[(size_t)(row0 + crow) * H_DIM + n0 + ccol] = acc[e];
    }
  }
}

// ---------------------------------------------------------------------------
// Kernel 3: persistent recurrent loop. One block = 16 batch rows, 4 waves.
// Wr kept resident in LDS (128KB of the 320KB/WGP budget); z in LDS bf16.
// ---------------------------------------------------------------------------
__global__ void __launch_bounds__(128)
alif_recurrent_kernel(const float* __restrict__ Icur,
                      const bf16* __restrict__ WrT,
                      const bf16* __restrict__ WoT,
                      const float* __restrict__ alpha,
                      const float* __restrict__ rho,
                      const float* __restrict__ oalpha,
                      const int*   __restrict__ y,
                      float* __restrict__ outputs,      // [T][B][OUT]
                      float* __restrict__ out_tail) {   // [loss, spikes]
  __shared__ bf16  wr_sh[H_DIM][WRPITCH];   // 135168 B
  __shared__ bf16  z_sh[16][WRPITCH];       //   8448 B
  __shared__ float ou_sh[16][33];           //   2112 B

  int tid  = threadIdx.x;
  int lane = tid & 31;
  int wave = tid >> 5;
  int bbase = blockIdx.x * 16;
  int arow = lane & 15;
  int hi   = lane >> 4;
  int ccol = lane & 15;

  // Stage Wr into LDS once (reused 1000 times).
  for (int i = tid; i < H_DIM * H_DIM; i += 128) {
    int k = i >> 8, h = i & 255;
    wr_sh[k][h] = WrT[i];
  }
  for (int i = tid; i < 16 * WRPITCH; i += 128) ((bf16*)z_sh)[i] = (bf16)0.0f;
  for (int i = tid; i < 16 * 33; i += 128) ((float*)ou_sh)[i] = 0.0f;

  v8f vzero = {};
  float al[4], rh[4];
  v8f u_st[4], a_st[4];
#pragma unroll
  for (int ti = 0; ti < 4; ++ti) {
    int h = (wave * 4 + ti) * 16 + ccol;
    al[ti] = alpha[h];
    rh[ti] = rho[h];
    u_st[ti] = vzero;
    a_st[ti] = vzero;
  }
  int ocol = wave * 16 + ccol;
  float oal = (wave < 2 && ocol < OUT_DIM) ? oalpha[ocol] : 0.0f;

  float spk_acc = 0.0f;
  float loss_acc = 0.0f;

  __syncthreads();

  for (int t = 0; t < T_STEPS; ++t) {
    // Prefetch next step's input current toward this WGP while we compute.
    if (t + 1 < T_STEPS) {
      const float* pf = &Icur[((size_t)(t + 1) * BSZ + bbase + (tid & 15)) * H_DIM
                              + (tid >> 4) * 32];
      __builtin_prefetch(pf, 0, 0);
    }

    // ---- Phase 1: rec = z_prev @ Wr^T  (A from z_sh, B from wr_sh) ----
    v8f rec[4];
#pragma unroll
    for (int ti = 0; ti < 4; ++ti) {
      int n0h = (wave * 4 + ti) * 16;
      v8f acc = {};
#pragma unroll
      for (int kc = 0; kc < 8; ++kc) {      // K = 256
        int kb = kc * 32;
        v16bf a, b;
#pragma unroll
        for (int e = 0; e < 16; ++e) {
          int vg = e >> 1, pos = e & 1;
          int k = kb + ((vg & 4) ? 16 : 0) + (hi ? 8 : 0) + ((vg & 3) << 1) + pos;
          a[e] = z_sh[arow][k];
        }
#pragma unroll
        for (int e = 0; e < 16; ++e)
          b[e] = wr_sh[kb + ((e & 8) ? 16 : 0) + (hi ? 8 : 0) + (e & 7)][n0h + ccol];
        acc = __builtin_amdgcn_wmma_f32_16x16x32_bf16(false, a, false, b,
                                                      (short)0, acc, false, false);
      }
      rec[ti] = acc;
    }
    __syncthreads();

    // ---- Phase 2: ALIF state update, write z_new into z_sh ----
#pragma unroll
    for (int ti = 0; ti < 4; ++ti) {
      int h = (wave * 4 + ti) * 16 + ccol;
      float a_ = al[ti], r_ = rh[ti];
#pragma unroll
      for (int e = 0; e < 8; ++e) {
        int rrow = (hi ? 8 : 0) + e;
        float cur = Icur[((size_t)t * BSZ + bbase + rrow) * H_DIM + h] + rec[ti][e];
        float zold = (float)z_sh[rrow][h];
        float u  = a_ * u_st[ti][e] + (1.0f - a_) * cur;
        float ad = r_ * a_st[ti][e] + (1.0f - r_) * zold;
        float theta = B0C + BETA * ad;
        float zn = (u - theta > 0.0f) ? 1.0f : 0.0f;
        u -= zn * theta;
        u_st[ti][e] = u;
        a_st[ti][e] = ad;
        z_sh[rrow][h] = (bf16)zn;
        spk_acc += zn;
      }
    }
    __syncthreads();

    // ---- Phase 3: readout  z_new @ Wo^T  (waves 0,1; WoT zero-padded) ----
    if (wave < 2) {
      int n0o = wave * 16;
      v8f acc = {};
#pragma unroll
      for (int kc = 0; kc < 8; ++kc) {
        int kb = kc * 32;
        v16bf a, b;
#pragma unroll
        for (int e = 0; e < 16; ++e) {
          int vg = e >> 1, pos = e & 1;
          int k = kb + ((vg & 4) ? 16 : 0) + (hi ? 8 : 0) + ((vg & 3) << 1) + pos;
          a[e] = z_sh[arow][k];
        }
#pragma unroll
        for (int e = 0; e < 16; ++e)          // k-strided gather, column n0o+ccol, no guard
          b[e] = WoT[(size_t)(kb + ((e & 8) ? 16 : 0) + (hi ? 8 : 0) + (e & 7)) * OPAD + n0o + ccol];
        acc = __builtin_amdgcn_wmma_f32_16x16x32_bf16(false, a, false, b,
                                                      (short)0, acc, false, false);
      }
      if (ocol < OUT_DIM) {
#pragma unroll
        for (int e = 0; e < 8; ++e) {
          int rrow = (hi ? 8 : 0) + e;
          float onew = oal * ou_sh[rrow][ocol] + (1.0f - oal) * acc[e];
          ou_sh[rrow][ocol] = onew;
          outputs[((size_t)t * BSZ + bbase + rrow) * OUT_DIM + ocol] = onew;
        }
      }
    }
    __syncthreads();

    // ---- Phase 4: per-step NLL(log_softmax), one thread per batch row ----
    // (next iteration's phase-1 barrier orders these reads vs. next ou writes)
    if (tid < 16) {
      int r = tid;
      float m = ou_sh[r][0];
#pragma unroll
      for (int o = 1; o < OUT_DIM; ++o) m = fmaxf(m, ou_sh[r][o]);
      float s = 0.0f;
#pragma unroll
      for (int o = 0; o < OUT_DIM; ++o) s += __expf(ou_sh[r][o] - m);
      int lbl = y[t * BSZ + bbase + r];
      loss_acc += ou_sh[r][lbl] - m - __logf(s);
    }
  }

  atomicAdd(&out_tail[1], spk_acc);
  if (tid < 16) atomicAdd(&out_tail[0], -loss_acc * (1.0f / (float)BSZ));
}

// ---------------------------------------------------------------------------
extern "C" void kernel_launch(void* const* d_in, const int* in_sizes, int n_in,
                              void* d_out, int out_size, void* d_ws, size_t ws_size,
                              hipStream_t stream) {
  const float* x       = (const float*)d_in[0];
  const int*   y       = (const int*)d_in[1];
  const float* Wh      = (const float*)d_in[2];
  const float* tau_mem = (const float*)d_in[3];
  const float* tau_adp = (const float*)d_in[4];
  const float* Wo      = (const float*)d_in[5];
  const float* out_tau = (const float*)d_in[6];

  float* out = (float*)d_out;
  float* out_tail = out + (size_t)T_STEPS * BSZ * OUT_DIM;  // [loss, spikes]

  char* ws = (char*)d_ws;
  size_t off = 0;
  bf16* WxT = (bf16*)(ws + off); off += (size_t)KPAD * H_DIM * sizeof(bf16);
  bf16* WrT = (bf16*)(ws + off); off += (size_t)H_DIM * H_DIM * sizeof(bf16);
  bf16* WoT = (bf16*)(ws + off); off += (size_t)H_DIM * OPAD * sizeof(bf16);
  off = (off + 15) & ~(size_t)15;
  float* alpha  = (float*)(ws + off); off += H_DIM * sizeof(float);
  float* rho    = (float*)(ws + off); off += H_DIM * sizeof(float);
  float* oalpha = (float*)(ws + off); off += (OUT_DIM * sizeof(float) + 15) & ~(size_t)15;
  float* Icur   = (float*)(ws + off);   // [T][B][H] f32 = 131 MB

  alif_init_kernel<<<512, 256, 0, stream>>>(Wh, Wo, tau_mem, tau_adp, out_tau,
                                            WxT, WrT, WoT, alpha, rho, oalpha, out_tail);
  alif_input_gemm_kernel<<<(T_STEPS * BSZ) / 16, 256, 0, stream>>>(x, WxT, Icur);
  alif_recurrent_kernel<<<BSZ / 16, 128, 0, stream>>>(Icur, WrT, WoT, alpha, rho,
                                                      oalpha, y, out, out_tail);
}
